// GraphGRUCell_12111807774828
// MI455X (gfx1250) — matrix operand order repaired and checked by reading, not compiled, not run on hardware
//
#include <hip/hip_runtime.h>
#include <hip/hip_bf16.h>
#include <math.h>

// ---- CDNA5 WMMA vector types -------------------------------------------------
typedef __attribute__((ext_vector_type(16))) __bf16 v16bf;
typedef __attribute__((ext_vector_type(8)))  float  v8f;

#define B_TOTAL   32768
#define R_N       4
#define U_N       256
#define D_N       256
#define COLS      768           // 3*U
#define NMAT      5             // [inputs@kernel, states[:,r]@rk[r] r=0..3]
#define KTILES    8             // 256 / 32
#define NTILES    48            // 768 / 16
#define FRAG_ELEMS 512          // 32 lanes * 16 bf16 per WMMA fragment
#define NFRAGS    (NMAT * KTILES * NTILES)          // 1920
#define WBUF_ELEMS ((size_t)NFRAGS * FRAG_ELEMS)    // 983040 bf16 each (hi, lo)
#define TS        3852          // padded LDS row stride in floats
#define LDS_S_FLOATS  (16 * TS)                      // 61632 floats
#define LDS_RED_FLOATS 64
// A-fragment staging region (bf16), after S+red:
#define AF_HI0 0
#define AF_LO0 4096
#define AF_HI1 8192
#define AF_LO1 12288            // total 16384 bf16 = 32 KB

// round-to-nearest-even f32 -> bf16 via bit ops
static __device__ inline __bf16 f2bf(float f) {
    unsigned u = __builtin_bit_cast(unsigned, f);
    unsigned r = (u + 0x7FFFu + ((u >> 16) & 1u)) >> 16;
    return __builtin_bit_cast(__bf16, (unsigned short)r);
}
static __device__ inline float bf2f(__bf16 h) {
    unsigned short s = __builtin_bit_cast(unsigned short, h);
    return __builtin_bit_cast(float, (unsigned)s << 16);
}

// ---- Kernel 1: rearrange weights into WMMA-fragment-major hi/lo bf16 ---------
// B-fragment k order used here (contiguous, same as A staging): lane 0-15 -> K=0..15, lane 16-31 -> K=16..31,
// element e -> K, column N = lane & 15.
__global__ __launch_bounds__(256) void prep_weights(
    const float* __restrict__ kernelW,    // (256, 768)
    const float* __restrict__ rk,         // (4, 256, 768)
    __bf16* __restrict__ whi, __bf16* __restrict__ wlo)
{
    int idx = blockIdx.x * 256 + threadIdx.x;           // 0 .. 983039
    int e    = idx & 15;
    int lane = (idx >> 4) & 31;
    int frag = idx >> 9;                                 // (m*8 + kt)*48 + nt
    int nt = frag % NTILES;
    int kt = (frag / NTILES) % KTILES;
    int m  = frag / (NTILES * KTILES);
    int n = nt * 16 + (lane & 15);
    int k = kt * 32 + (lane >> 4) * 16 + e;
    float w = (m == 0) ? kernelW[k * COLS + n]
                       : rk[(((m - 1) * D_N) + k) * COLS + n];
    __bf16 h = f2bf(w);
    whi[idx] = h;
    wlo[idx] = f2bf(w - bf2f(h));
}

// ---- elementwise GRU tail for one 16-row half --------------------------------
static __device__ inline void gru_elementwise(
    const float* __restrict__ S, float* __restrict__ red,
    const float* __restrict__ states, const unsigned char* __restrict__ cell_mask,
    const float* __restrict__ v, float* __restrict__ out,
    int bbase, int tid, int lane, int wave)
{
    const int u = tid;                 // 0..255 == U
    const float vu = v[u];
    for (int row = 0; row < 16; ++row) {
        const int b = bbase + row;
        const float* Srow = S + row * TS;

        const float xz = Srow[u];
        const float xr = Srow[U_N + u];
        const float xh = Srow[2 * U_N + u];

        float rz[4];
        float acc_h = 0.f;
#pragma unroll
        for (int r = 0; r < R_N; ++r) {
            const float* Sm = Srow + (r + 1) * COLS;
            rz[r]     = Sm[u];
            float rr  = Sm[U_N + u];
            float rhc = Sm[2 * U_N + u];
            bool mk = (cell_mask[b * R_N + r] != 0);
            float sg = 1.f / (1.f + __expf(-(xr + rr)));
            acc_h += mk ? sg * rhc : 0.f;
        }
        const float hh = tanhf(xh + acc_h * 0.25f);

        float part[5];
#pragma unroll
        for (int k = 0; k < 4; ++k) part[k] = tanhf(xz + rz[k]) * vu;
        part[4] = tanhf(xz + hh) * vu;

#pragma unroll
        for (int k = 0; k < 5; ++k) {
            float s = part[k];
#pragma unroll
            for (int off = 16; off > 0; off >>= 1) s += __shfl_down(s, off, 32);
            if (lane == 0) red[wave * 5 + k] = s;
        }
        __syncthreads();
        if (tid == 0) {
            float lg[5];
#pragma unroll
            for (int k = 0; k < 5; ++k) {
                float s = 0.f;
                for (int w = 0; w < 8; ++w) s += red[w * 5 + k];
                lg[k] = s;
            }
            for (int r = 0; r < R_N; ++r)
                if (cell_mask[b * R_N + r] == 0) lg[r] = -__builtin_inff();
            float mx = lg[4];
            for (int k = 0; k < 4; ++k) mx = fmaxf(mx, lg[k]);
            float den = 0.f, ex[5];
            for (int k = 0; k < 5; ++k) { ex[k] = __expf(lg[k] - mx); den += ex[k]; }
            for (int k = 0; k < 5; ++k) red[40 + k] = ex[k] / den;
        }
        __syncthreads();

        float prob[5];
#pragma unroll
        for (int k = 0; k < 5; ++k) prob[k] = red[40 + k];

        float h = prob[4] * hh;   // output bank: [states, hh]
#pragma unroll
        for (int r = 0; r < R_N; ++r)
            h += prob[r] * states[((size_t)b * R_N + r) * U_N + u];
        out[(size_t)b * U_N + u] = h;
        __syncthreads();          // red reused next row
    }
}

// ---- Kernel 2: fused 5-way GEMM (32 rows/block, bf16 hi/lo WMMA) + GRU -------
__global__ __launch_bounds__(256, 1) void graphgru_fused(
    const float* __restrict__ inputs,            // (B, 256)
    const float* __restrict__ states,            // (B, 4, 256)
    const unsigned char* __restrict__ cell_mask, // (B, 4) bool bytes
    const float* __restrict__ bias,              // (5, 768)
    const float* __restrict__ v,                 // (256,)
    const __bf16* __restrict__ whi,
    const __bf16* __restrict__ wlo,
    float* __restrict__ out)                     // (B, 256)
{
    extern __shared__ float smem[];
    float*  S   = smem;                               // [16][TS]
    float*  red = smem + LDS_S_FLOATS;                // 64 floats
    __bf16* Afr = (__bf16*)(smem + LDS_S_FLOATS + LDS_RED_FLOATS);  // 16384 bf16

    const int tid  = threadIdx.x;
    const int lane = tid & 31;
    const int wave = tid >> 5;
    const int b0   = blockIdx.x * 32;

    v8f acc1s[NMAT][6];     // tile1 (rows 16..31) accumulators, persistent

    // ---------------- GEMM phase ----------------
#pragma unroll
    for (int m = 0; m < NMAT; ++m) {
        __syncthreads();    // previous m's A-fragment reads complete
        // Stage A hi/lo fragments for BOTH row tiles: thread t <-> (kt = t>>5, l = t&31)
        {
            const int kt   = tid >> 5;
            const int l    = tid & 31;
            const int rowl = l & 15;
            const int sl   = l >> 4;
            const int k0   = kt * 32 + sl * 16;
            const int rstride = (m == 0) ? D_N : (R_N * U_N);
            const float* a0 = ((m == 0)
                ? (inputs + (size_t)(b0 + rowl) * D_N)
                : (states + ((size_t)(b0 + rowl) * R_N + (m - 1)) * U_N)) + k0;
            const float* a1 = a0 + (size_t)16 * rstride;   // rows +16
            const int slot = (kt * 32 + l) * 16;
#pragma unroll
            for (int j = 0; j < 8; ++j) {
                float x0 = a0[j],      y0 = a0[8 + j];
                float x1 = a1[j],      y1 = a1[8 + j];
                __bf16 hx0 = f2bf(x0), hy0 = f2bf(y0);
                __bf16 hx1 = f2bf(x1), hy1 = f2bf(y1);
                Afr[AF_HI0 + slot + j]     = hx0;
                Afr[AF_HI0 + slot + 8 + j] = hy0;
                Afr[AF_LO0 + slot + j]     = f2bf(x0 - bf2f(hx0));
                Afr[AF_LO0 + slot + 8 + j] = f2bf(y0 - bf2f(hy0));
                Afr[AF_HI1 + slot + j]     = hx1;
                Afr[AF_HI1 + slot + 8 + j] = hy1;
                Afr[AF_LO1 + slot + j]     = f2bf(x1 - bf2f(hx1));
                Afr[AF_LO1 + slot + 8 + j] = f2bf(y1 - bf2f(hy1));
            }
        }
        __syncthreads();

        v8f acc0[6];
        v8f zero = {0.f, 0.f, 0.f, 0.f, 0.f, 0.f, 0.f, 0.f};
#pragma unroll
        for (int t = 0; t < 6; ++t) { acc0[t] = zero; acc1s[m][t] = zero; }

        for (int kt = 0; kt < KTILES; ++kt) {
            const int slot = (kt * 32 + lane) * 16;
            v16bf ahi0 = *reinterpret_cast<const v16bf*>(Afr + AF_HI0 + slot);
            v16bf alo0 = *reinterpret_cast<const v16bf*>(Afr + AF_LO0 + slot);
            v16bf ahi1 = *reinterpret_cast<const v16bf*>(Afr + AF_HI1 + slot);
            v16bf alo1 = *reinterpret_cast<const v16bf*>(Afr + AF_LO1 + slot);
#pragma unroll
            for (int t = 0; t < 6; ++t) {
                const int nt = wave * 6 + t;
                const size_t fo = (((size_t)(m * KTILES + kt)) * NTILES + nt) * FRAG_ELEMS
                                  + (size_t)lane * 16;
                v16bf bhi = *reinterpret_cast<const v16bf*>(whi + fo);
                v16bf blo = *reinterpret_cast<const v16bf*>(wlo + fo);
                // B loaded once, feeds both row tiles (halves weight traffic)
                acc0[t] = __builtin_amdgcn_wmma_f32_16x16x32_bf16(
                              false, ahi0, false, bhi, (short)0, acc0[t], false, false);
                acc0[t] = __builtin_amdgcn_wmma_f32_16x16x32_bf16(
                              false, ahi0, false, blo, (short)0, acc0[t], false, false);
                acc0[t] = __builtin_amdgcn_wmma_f32_16x16x32_bf16(
                              false, alo0, false, bhi, (short)0, acc0[t], false, false);
                acc1s[m][t] = __builtin_amdgcn_wmma_f32_16x16x32_bf16(
                              false, ahi1, false, bhi, (short)0, acc1s[m][t], false, false);
                acc1s[m][t] = __builtin_amdgcn_wmma_f32_16x16x32_bf16(
                              false, ahi1, false, blo, (short)0, acc1s[m][t], false, false);
                acc1s[m][t] = __builtin_amdgcn_wmma_f32_16x16x32_bf16(
                              false, alo1, false, bhi, (short)0, acc1s[m][t], false, false);
            }
        }
        // Stage tile0 (+bias) into S. C layout: VGPR i -> M = (lane>>4)*8 + i, N = lane&15.
#pragma unroll
        for (int t = 0; t < 6; ++t) {
            const int nt  = wave * 6 + t;
            const int c   = nt * 16 + (lane & 15);
            const float bv = bias[m * COLS + c];
            const int col = m * COLS + c;
#pragma unroll
            for (int i = 0; i < 8; ++i) {
                const int rowM = (lane >> 4) * 8 + i;
                S[rowM * TS + col] = acc0[t][i] + bv;
            }
        }
    }
    __syncthreads();

    // ---------------- elementwise: rows b0 .. b0+15 ----------------
    gru_elementwise(S, red, states, cell_mask, v, out, b0, tid, lane, wave);
    __syncthreads();

    // ---------------- stage tile1 (+bias) into S, then rows b0+16 .. b0+31 ----
#pragma unroll
    for (int m = 0; m < NMAT; ++m) {
#pragma unroll
        for (int t = 0; t < 6; ++t) {
            const int nt  = wave * 6 + t;
            const int c   = nt * 16 + (lane & 15);
            const float bv = bias[m * COLS + c];
            const int col = m * COLS + c;
#pragma unroll
            for (int i = 0; i < 8; ++i) {
                const int rowM = (lane >> 4) * 8 + i;
                S[rowM * TS + col] = acc1s[m][t][i] + bv;
            }
        }
    }
    __syncthreads();
    gru_elementwise(S, red, states, cell_mask, v, out, b0 + 16, tid, lane, wave);
}

// ---- host launch -------------------------------------------------------------
extern "C" void kernel_launch(void* const* d_in, const int* in_sizes, int n_in,
                              void* d_out, int out_size, void* d_ws, size_t ws_size,
                              hipStream_t stream) {
    const float*         inputs    = (const float*)d_in[0];
    const float*         states    = (const float*)d_in[1];
    /* d_in[2] edge_types: dead in reference */
    const unsigned char* cell_mask = (const unsigned char*)d_in[3];
    const float*         kernelW   = (const float*)d_in[4];
    const float*         rk        = (const float*)d_in[5];
    const float*         bias      = (const float*)d_in[6];
    const float*         v         = (const float*)d_in[7];
    /* d_in[8] edge_emb: dead in reference */
    float* out = (float*)d_out;

    __bf16* whi = (__bf16*)d_ws;
    __bf16* wlo = whi + WBUF_ELEMS;

    const size_t ldsBytes = (size_t)(LDS_S_FLOATS + LDS_RED_FLOATS) * sizeof(float)
                          + (size_t)16384 * sizeof(__bf16);   // ~279.5 KB < 320 KB/WGP
    (void)hipFuncSetAttribute(reinterpret_cast<const void*>(graphgru_fused),
                              hipFuncAttributeMaxDynamicSharedMemorySize, (int)ldsBytes);

    prep_weights<<<(int)(WBUF_ELEMS / 256), 256, 0, stream>>>(kernelW, rk, whi, wlo);
    graphgru_fused<<<B_TOTAL / 32, 256, ldsBytes, stream>>>(
        inputs, states, cell_mask, bias, v, whi, wlo, out);
}
